// ClassificationModel_2894807957808
// MI455X (gfx1250) — compile-verified, matrix operand order checked by reading, not run on hardware
//
#include <hip/hip_runtime.h>

// ============================================================================
// DGCNN classifier for MI455X (gfx1250, wave32, WMMA + async-LDS).
//
// All heavy math runs on v_wmma_f32_16x16x32_f16 (f16 in, f32 accumulate):
//   - per-batch Gram matrix for dynamic kNN (streamed in 128-col chunks)
//   - EdgeConv refactored to one GEMM per layer:  F = X * [U ; V]^T
//     with U = W[:, :C], V = W[:, C:] - W[:, :C]  (W*[xj-xi;xi] = U xj + V xi)
//   - W5 projection: workgroup-cooperative GEMM, B strip staged into LDS with
//     GLOBAL_LOAD_ASYNC_TO_LDS_B128 + s_wait_asynccnt, fused BN5 + LeakyReLU +
//     global max-pool (sign-aware float atomicMax)
//
// f16 operands are staged zero-padded to K%32==0 so WMMA fragment loads are
// two unconditional 16-byte loads per lane (global_load_b128 / ds_load_b128).
// Each wave computes a 16x64 strip: 1 A fragment x 4 B fragments -> 4 WMMA.
// ============================================================================

#define BATCH   8
#define NPTS    2048
#define BNTOT   (BATCH * NPTS)   // 16384
#define KNN     10
#define CCAT    512
#define CHUNKC  128              // distance-matrix column chunk
#define SLOPE_F 0.2f
#define EPS_BN  1e-5f
#define TKW5    256              // K sub-strip staged in LDS for the W5 GEMM

typedef __attribute__((ext_vector_type(16))) _Float16 v16h;
typedef __attribute__((ext_vector_type(8)))  _Float16 v8h;
typedef __attribute__((ext_vector_type(8)))  float    v8f;

__device__ __forceinline__ float lrelu(float v) { return v < 0.0f ? SLOPE_F * v : v; }

// 16-bit WMMA A/B fragment loader (wave32), row-major source, K padded to 32.
// ISA layout: lanes 0-15 -> kbase 0, lanes 16-31 -> kbase 8; a lane's 16
// elements are K = kbase+0..7 and kbase+16..23 -> two contiguous b128 loads.
__device__ __forceinline__ v16h load_frag16(const _Float16* rowp, int k0) {
  const int koff = (threadIdx.x & 16) ? 8 : 0;
  const _Float16* p = rowp + k0 + koff;
  v8h lo = *(const v8h*)(p);
  v8h hi = *(const v8h*)(p + 16);
  return __builtin_shufflevector(lo, hi, 0, 1, 2, 3, 4, 5, 6, 7,
                                 8, 9, 10, 11, 12, 13, 14, 15);
}

#define WMMA_F16(A, B, C) \
  __builtin_amdgcn_wmma_f32_16x16x32_f16(false, (A), false, (B), (short)0, (C), false, false)

// ---------------------------------------------------------------------------
// prep: strided f32 features -> compact f16 rows (zero-padded to Kp) + sq norms
__global__ void prep_kernel(const float* __restrict__ x, int ldx, int C, int Kp,
                            _Float16* __restrict__ xh, float* __restrict__ x2) {
  int bn = blockIdx.x * blockDim.x + threadIdx.x;
  if (bn >= BNTOT) return;
  const float* xr = x + (size_t)bn * ldx;
  _Float16* hr = xh + (size_t)bn * Kp;
  float s = 0.0f;
  for (int c = 0; c < Kp; ++c) {
    float v = (c < C) ? xr[c] : 0.0f;
    s += v * v;
    hr[c] = (_Float16)v;
  }
  x2[bn] = s;
}

// build fused EdgeConv weight [U ; V] (2*Cout x Kp, zero-padded) in f16
__global__ void wcat_kernel(const float* __restrict__ W, _Float16* __restrict__ out,
                            int Cout, int Cin, int Kp) {
  int i = blockIdx.x * blockDim.x + threadIdx.x;
  if (i >= 2 * Cout * Kp) return;
  int o2 = i / Kp, c = i % Kp;
  float v = 0.0f;
  if (c < Cin) {
    if (o2 < Cout) {
      v = W[(size_t)o2 * 2 * Cin + c];
    } else {
      int o = o2 - Cout;
      v = W[(size_t)o * 2 * Cin + Cin + c] - W[(size_t)o * 2 * Cin + c];
    }
  }
  out[i] = (_Float16)v;
}

__global__ void f2h_kernel(const float* __restrict__ src, _Float16* __restrict__ dst, int n) {
  int i = blockIdx.x * blockDim.x + threadIdx.x;
  if (i < n) dst[i] = (_Float16)src[i];
}

__global__ void neg_inf_init_kernel(float* __restrict__ p, int n) {
  int i = blockIdx.x * blockDim.x + threadIdx.x;
  if (i < n) p[i] = -__builtin_inff();
}

// ---------------------------------------------------------------------------
// Gram/dist strip: one wave -> 16x64 tile of D = x2_i - 2*xi.xj + x2_j (self=inf)
// grid: (NPTS/16, CHUNKC/64, BATCH), block: 32
__global__ void dist_tile_kernel(const _Float16* __restrict__ xh, const float* __restrict__ x2,
                                 float* __restrict__ dist, int Kp, int col0) {
  const int lane = threadIdx.x & 31;
  const int m = lane & 15;
  const int row0 = blockIdx.x * 16;
  const int c0 = col0 + blockIdx.y * 64;
  const int b = blockIdx.z;
  const _Float16* Ar = xh + ((size_t)b * NPTS + row0 + m) * Kp;
  const _Float16* Br = xh + ((size_t)b * NPTS + c0 + m) * Kp;
  const size_t bstep = (size_t)16 * Kp;
  v8f acc[4] = {{}, {}, {}, {}};
  for (int k0 = 0; k0 < Kp; k0 += 32) {
    __builtin_prefetch(Ar + k0 + 256, 0, 3);
    v16h a = load_frag16(Ar, k0);
#pragma unroll
    for (int s = 0; s < 4; ++s) {
      v16h bf = load_frag16(Br + s * bstep, k0);
      acc[s] = WMMA_F16(a, bf, acc[s]);
    }
  }
  const int rbase = row0 + ((lane & 16) ? 8 : 0);
#pragma unroll
  for (int s = 0; s < 4; ++s) {
    const int j = c0 + s * 16 + m;
    const float x2j = x2[(size_t)b * NPTS + j];
#pragma unroll
    for (int r = 0; r < 8; ++r) {
      const int i = rbase + r;
      float d = x2[(size_t)b * NPTS + i] + x2j - 2.0f * acc[s][r];
      if (i == j) d = __builtin_inff();          // drop self
      dist[((size_t)b * NPTS + i) * CHUNKC + (j - col0)] = d;
    }
  }
}

// streaming top-10 (smallest) update over one 128-column chunk; thread per point
__global__ void topk_update_kernel(const float* __restrict__ dist, float* __restrict__ tv,
                                   int* __restrict__ ti, int col0, int first) {
  int bn = blockIdx.x * blockDim.x + threadIdx.x;
  if (bn >= BNTOT) return;
  int b = bn / NPTS;
  float v[KNN]; int id[KNN];
  if (first) {
#pragma unroll
    for (int t = 0; t < KNN; ++t) { v[t] = __builtin_inff(); id[t] = 0; }
  } else {
#pragma unroll
    for (int t = 0; t < KNN; ++t) {
      v[t] = tv[(size_t)bn * KNN + t]; id[t] = ti[(size_t)bn * KNN + t];
    }
  }
  const float* dr = dist + (size_t)bn * CHUNKC;
  for (int c = 0; c < CHUNKC; ++c) {
    float d = dr[c];
    if (d < v[KNN - 1]) {
      int jid = b * NPTS + col0 + c;             // absolute neighbor row
      int t = KNN - 1;
      while (t > 0 && v[t - 1] > d) { v[t] = v[t - 1]; id[t] = id[t - 1]; --t; }
      v[t] = d; id[t] = jid;
    }
  }
#pragma unroll
  for (int t = 0; t < KNN; ++t) {
    tv[(size_t)bn * KNN + t] = v[t]; ti[(size_t)bn * KNN + t] = id[t];
  }
}

// ---------------------------------------------------------------------------
// WMMA GEMM strip: Y[M x Nout] = Xh[M x Kp] * Wh[Nout x Kp]^T  (16x64 per wave)
// grid: (M/16, Nout/64), block: 32
__global__ void gemm_xwT_kernel(const _Float16* __restrict__ Xh,
                                const _Float16* __restrict__ Wh,
                                float* __restrict__ Y, int ldy, int Kp) {
  const int lane = threadIdx.x & 31;
  const int m = lane & 15;
  const int row0 = blockIdx.x * 16;
  const int n0 = blockIdx.y * 64;
  const _Float16* Ar = Xh + (size_t)(row0 + m) * Kp;
  const _Float16* Br = Wh + (size_t)(n0 + m) * Kp;
  const size_t bstep = (size_t)16 * Kp;
  v8f acc[4] = {{}, {}, {}, {}};
  for (int k0 = 0; k0 < Kp; k0 += 32) {
    __builtin_prefetch(Ar + k0 + 256, 0, 3);
    v16h a = load_frag16(Ar, k0);
#pragma unroll
    for (int s = 0; s < 4; ++s) {
      v16h bf = load_frag16(Br + s * bstep, k0);
      acc[s] = WMMA_F16(a, bf, acc[s]);
    }
  }
  const int rbase = row0 + ((lane & 16) ? 8 : 0);
#pragma unroll
  for (int s = 0; s < 4; ++s) {
    const int col = n0 + s * 16 + m;
#pragma unroll
    for (int r = 0; r < 8; ++r) Y[(size_t)(rbase + r) * ldy + col] = acc[s][r];
  }
}

__device__ __forceinline__ void atomic_max_f32(float* addr, float v) {
  if (v >= 0.0f) atomicMax((int*)addr, __float_as_int(v));
  else           atomicMin((unsigned int*)addr, __float_as_uint(v));
}

// ---------------------------------------------------------------------------
// W5 stage: h = LReLU(BN5(Xcat * W5^T)), fused max over points -> g[b][1024].
// Workgroup-cooperative: 8 waves, 128x64 tile; the 64-row weight strip is
// staged into LDS in 256-K sub-strips with async global->LDS copies
// (global_load_async_to_lds_b128 + s_wait_asynccnt), then each wave runs the
// 1x4 WMMA strip against LDS-resident B fragments.
// grid: (BNTOT/128, 1024/64), block: 256
__global__ __launch_bounds__(256) void gemm_w5_fused_kernel(
    const _Float16* __restrict__ Xh, const _Float16* __restrict__ Wh,
    const float* __restrict__ gamma, const float* __restrict__ beta,
    const float* __restrict__ mean,  const float* __restrict__ var,
    float* __restrict__ g) {
  __shared__ _Float16 smem[64 * TKW5];           // 32 KB weight sub-strip
  const int lane = threadIdx.x & 31;
  const int wave = threadIdx.x >> 5;
  const int m = lane & 15;
  const int row0 = blockIdx.x * 128 + wave * 16;
  const int n0 = blockIdx.y * 64;
  const _Float16* Ar = Xh + (size_t)(row0 + m) * CCAT;
  // generic->LDS: per ISA aperture rule, LDS byte offset = addr[31:0]
  const unsigned lds_base = (unsigned)(uintptr_t)&smem[0];
  const char* wbase = (const char*)Wh;
  v8f acc[4] = {{}, {}, {}, {}};

  for (int kt = 0; kt < CCAT; kt += TKW5) {
    // ---- async stage: 64 rows x TKW5 cols (2 bytes each) -> LDS ----
#pragma unroll
    for (int it = 0; it < (64 * TKW5 * 2) / (256 * 16); ++it) {   // 8 iters
      const int off = threadIdx.x * 16 + it * 4096;               // strip byte
      const int row = off / (TKW5 * 2);
      const int rcol = off % (TKW5 * 2);
      unsigned long long gaddr =
          (unsigned long long)(uintptr_t)(wbase + (size_t)(n0 + row) * (CCAT * 2)
                                          + (size_t)kt * 2 + rcol);
      unsigned laddr = lds_base + (unsigned)off;
      asm volatile("global_load_async_to_lds_b128 %0, %1, off"
                   :: "v"(laddr), "v"(gaddr) : "memory");
    }
    asm volatile("s_wait_asynccnt 0x0" ::: "memory");
    __syncthreads();
    // ---- compute: 1 A fragment x 4 LDS B fragments per 32-K chunk ----
    for (int k0 = 0; k0 < TKW5; k0 += 32) {
      __builtin_prefetch(Ar + kt + k0 + 256, 0, 3);
      v16h a = load_frag16(Ar, kt + k0);
#pragma unroll
      for (int s = 0; s < 4; ++s) {
        v16h bf = load_frag16(&smem[(s * 16 + m) * TKW5], k0);
        acc[s] = WMMA_F16(a, bf, acc[s]);
      }
    }
    __syncthreads();
  }

  const int b = row0 / NPTS;                     // 128-row tile stays in one batch
#pragma unroll
  for (int s = 0; s < 4; ++s) {
    const int col = n0 + s * 16 + m;
    const float sc = gamma[col] * rsqrtf(var[col] + EPS_BN);
    const float bb = beta[col] - mean[col] * sc;
    float mx = -__builtin_inff();
#pragma unroll
    for (int r = 0; r < 8; ++r) mx = fmaxf(mx, lrelu(sc * acc[s][r] + bb));
    // rows 0-7 live in lanes 0-15, rows 8-15 in lanes 16-31 (same column)
    mx = fmaxf(mx, __shfl_xor(mx, 16, 32));
    if (lane < 16) atomic_max_f32(&g[b * 1024 + col], mx);
  }
}

// ---------------------------------------------------------------------------
// EdgeConv epilogue: max over 10 neighbors of (U x_j) + (V x_i), then BN+LReLU.
// Sign-aware: max_k LReLU(s*h+b) uses max h for s>=0, min h for s<0.
// grid: BNTOT blocks, block: Cout threads
__global__ void edge_max_kernel(const float* __restrict__ F, int ldF,
                                const int* __restrict__ ti,
                                const float* __restrict__ gamma, const float* __restrict__ beta,
                                const float* __restrict__ mean,  const float* __restrict__ var,
                                float* __restrict__ xcat, int off, int Cout) {
  const int o = threadIdx.x;
  const int bn = blockIdx.x;
  const float self = F[(size_t)bn * ldF + Cout + o];   // V * x_i
  float mx = -__builtin_inff(), mn = __builtin_inff();
#pragma unroll
  for (int t = 0; t < KNN; ++t) {
    const int nb = ti[(size_t)bn * KNN + t];
    const float f = F[(size_t)nb * ldF + o];           // U * x_j
    mx = fmaxf(mx, f); mn = fminf(mn, f);
  }
  const float s  = gamma[o] * rsqrtf(var[o] + EPS_BN);
  const float bb = beta[o] - mean[o] * s;
  float h = ((s >= 0.0f) ? mx : mn) + self;
  h = lrelu(s * h + bb);
  xcat[(size_t)bn * CCAT + off + o] = h;
}

// small dense FC: out[b,o] = act(BN(X[b,:] . W[o,:] + bias))
__global__ void fc_kernel(const float* __restrict__ X, const float* __restrict__ W,
                          const float* __restrict__ bias,
                          const float* __restrict__ gamma, const float* __restrict__ beta,
                          const float* __restrict__ mean,  const float* __restrict__ var,
                          float* __restrict__ out, int Bb, int K, int O, int act) {
  int i = blockIdx.x * blockDim.x + threadIdx.x;
  if (i >= Bb * O) return;
  int b = i / O, o = i % O;
  const float* xr = X + (size_t)b * K;
  const float* wr = W + (size_t)o * K;
  float s = 0.0f;
  for (int k = 0; k < K; ++k) s += xr[k] * wr[k];
  if (bias)  s += bias[o];
  if (gamma) { float sc = gamma[o] * rsqrtf(var[o] + EPS_BN); s = sc * s + (beta[o] - mean[o] * sc); }
  if (act)   s = lrelu(s);
  out[i] = s;
}

// ============================================================================
extern "C" void kernel_launch(void* const* d_in, const int* in_sizes, int n_in,
                              void* d_out, int out_size, void* d_ws, size_t ws_size,
                              hipStream_t stream) {
  (void)in_sizes; (void)n_in; (void)out_size; (void)ws_size;

  // ---- inputs (setup_inputs dict order, nested dicts flattened in order) ----
  const float* points = (const float*)d_in[0];
  const float* Wl[4]  = { (const float*)d_in[1], (const float*)d_in[2],
                          (const float*)d_in[3], (const float*)d_in[4] };
  const float* W5     = (const float*)d_in[5];
  // bn1..bn5 each {gamma,beta,mean,var} at 6,10,14,18,22
  const int bnidx[5]  = { 6, 10, 14, 18, 22 };
  const float* fw1    = (const float*)d_in[26];
  const int fcbn1     = 27;
  const float* fw2    = (const float*)d_in[31];
  const float* fb2    = (const float*)d_in[32];
  const int fcbn2     = 33;
  const float* fw3    = (const float*)d_in[37];
  const float* fb3    = (const float*)d_in[38];

  // ---- workspace layout ----
  char* ws = (char*)d_ws;
  size_t off = 0;
  auto alloc = [&](size_t bytes) { void* p = ws + off; off = (off + bytes + 255) & ~(size_t)255; return p; };
  _Float16* xh   = (_Float16*)alloc((size_t)BNTOT * CCAT * 2);   // f16 features (Kp-padded)
  float*    x2   = (float*)   alloc((size_t)BNTOT * 4);          // squared norms
  _Float16* wch  = (_Float16*)alloc((size_t)2 * 256 * 256 * 2);  // fused edge weights
  _Float16* w5h  = (_Float16*)alloc((size_t)1024 * 512 * 2);     // W5 f16
  float*    F    = (float*)   alloc((size_t)BNTOT * 512 * 4);    // edge GEMM out
  float*    dist = (float*)F;                                    // chunk dist (aliases F; disjoint lifetime)
  float*    tv   = (float*)   alloc((size_t)BNTOT * KNN * 4);
  int*      ti   = (int*)     alloc((size_t)BNTOT * KNN * 4);
  float*    xcat = (float*)   alloc((size_t)BNTOT * CCAT * 4);   // concat features
  float*    g    = (float*)   alloc((size_t)BATCH * 1024 * 4);   // pooled global feature
  float*    fc1o = (float*)   alloc((size_t)BATCH * 512 * 4);
  float*    fc2o = (float*)   alloc((size_t)BATCH * 256 * 4);

  // ---- layer configs ----
  const int cin[4]  = { 3, 64, 64, 128 };
  const int kp[4]   = { 32, 64, 64, 128 };   // K padded to multiple of 32
  const int cout[4] = { 64, 64, 128, 256 };
  const int coff[4] = { 0, 64, 128, 256 };
  const float* lin[4] = { points, xcat + 0, xcat + 64, xcat + 128 };
  const int lld[4]    = { 3, CCAT, CCAT, CCAT };

  for (int l = 0; l < 4; ++l) {
    const int Ci = cin[l], Co = cout[l], Kp = kp[l];
    const float* bg = (const float*)d_in[bnidx[l] + 0];
    const float* bb = (const float*)d_in[bnidx[l] + 1];
    const float* bm = (const float*)d_in[bnidx[l] + 2];
    const float* bv = (const float*)d_in[bnidx[l] + 3];

    prep_kernel<<<(BNTOT + 255) / 256, 256, 0, stream>>>(lin[l], lld[l], Ci, Kp, xh, x2);
    wcat_kernel<<<(2 * Co * Kp + 255) / 256, 256, 0, stream>>>(Wl[l], wch, Co, Ci, Kp);

    // streamed kNN: WMMA Gram strips -> dist chunk -> running top-10
    for (int c = 0; c < NPTS / CHUNKC; ++c) {
      dist_tile_kernel<<<dim3(NPTS / 16, CHUNKC / 64, BATCH), 32, 0, stream>>>(
          xh, x2, dist, Kp, c * CHUNKC);
      topk_update_kernel<<<(BNTOT + 127) / 128, 128, 0, stream>>>(
          dist, tv, ti, c * CHUNKC, c == 0 ? 1 : 0);
    }

    // EdgeConv GEMM: F = X * [U;V]^T  (BNTOT x 2*Co)
    gemm_xwT_kernel<<<dim3(BNTOT / 16, (2 * Co) / 64), 32, 0, stream>>>(
        xh, wch, F, 2 * Co, Kp);

    // gather neighbors, max over k, BN + LeakyReLU -> concat buffer
    edge_max_kernel<<<BNTOT, Co, 0, stream>>>(F, 2 * Co, ti, bg, bb, bm, bv,
                                              xcat, coff[l], Co);
  }

  // ---- global stage: W5 GEMM fused with BN5 + LReLU + max-pool over points ----
  prep_kernel<<<(BNTOT + 255) / 256, 256, 0, stream>>>(xcat, CCAT, CCAT, CCAT, xh, x2);
  f2h_kernel<<<(1024 * 512 + 255) / 256, 256, 0, stream>>>(W5, w5h, 1024 * 512);
  neg_inf_init_kernel<<<(BATCH * 1024 + 255) / 256, 256, 0, stream>>>(g, BATCH * 1024);
  {
    const float* bg = (const float*)d_in[bnidx[4] + 0];
    const float* bb = (const float*)d_in[bnidx[4] + 1];
    const float* bm = (const float*)d_in[bnidx[4] + 2];
    const float* bv = (const float*)d_in[bnidx[4] + 3];
    gemm_w5_fused_kernel<<<dim3(BNTOT / 128, 1024 / 64), 256, 0, stream>>>(
        xh, w5h, bg, bb, bm, bv, g);
  }

  // ---- FC head ----
  fc_kernel<<<(BATCH * 512 + 63) / 64, 64, 0, stream>>>(
      g, fw1, nullptr,
      (const float*)d_in[fcbn1 + 0], (const float*)d_in[fcbn1 + 1],
      (const float*)d_in[fcbn1 + 2], (const float*)d_in[fcbn1 + 3],
      fc1o, BATCH, 1024, 512, 1);
  fc_kernel<<<(BATCH * 256 + 63) / 64, 64, 0, stream>>>(
      fc1o, fw2, fb2,
      (const float*)d_in[fcbn2 + 0], (const float*)d_in[fcbn2 + 1],
      (const float*)d_in[fcbn2 + 2], (const float*)d_in[fcbn2 + 3],
      fc2o, BATCH, 512, 256, 1);
  fc_kernel<<<(BATCH * 3 + 63) / 64, 64, 0, stream>>>(
      fc2o, fw3, fb3, nullptr, nullptr, nullptr, nullptr,
      (float*)d_out, BATCH, 256, 3, 0);
}
